// Model_44985487458696
// MI455X (gfx1250) — matrix operand order repaired and checked by reading, not compiled, not run on hardware
//
#include <hip/hip_runtime.h>
#include <hip/hip_bf16.h>

typedef float     v8f  __attribute__((ext_vector_type(8)));
typedef _Float16  v8h  __attribute__((ext_vector_type(8)));
typedef _Float16  v16h __attribute__((ext_vector_type(16)));

// ---------- helpers ----------
static __device__ __forceinline__ unsigned encf(float x) {
    unsigned u = __float_as_uint(x);
    return (u & 0x80000000u) ? ~u : (u | 0x80000000u);   // monotone float->uint
}
static __device__ __forceinline__ float decf(unsigned u) {
    return __uint_as_float((u & 0x80000000u) ? (u & 0x7FFFFFFFu) : ~u);
}
static __device__ __forceinline__ float lrelu02(float x) { return x > 0.f ? x : 0.2f * x; }

static __device__ __forceinline__ void atomAddF(float* p, float v) {
    unsafeAtomicAdd(p, v);   // lowers to global_atomic_add_f32 on gfx1250
}

// Build 16-bit A fragment (16x32 MxK tile), contiguous k order:
// lanes 0-15 : K0..K15, lanes 16-31 : K16..K31
static __device__ __forceinline__ v16h make_afrag(const _Float16* ap) {
    v8h a0 = *(const v8h*)ap;
    v8h a1 = *(const v8h*)(ap + 8);
    v16h r;
#pragma unroll
    for (int i = 0; i < 8; ++i) { r[i] = a0[i]; r[i + 8] = a1[i]; }
    return r;
}

// ---------------------------------------------------------------------------
// Generic tiled GEMM:  C[nrows,NC] = A[nrows,K] @ W[K,NC] (+bias) (+relu)
// optional fused attention logits: o_asrc[i] = C_row_i . aw_src ; same for dst
// Block: 256 threads = 8 waves, each wave owns a 16-row tile (block: 128 rows).
// ---------------------------------------------------------------------------
template <int K, int NC, bool ALPHA>
__global__ __launch_bounds__(256) void gemm_tiled(
    const float* __restrict__ A, const float* __restrict__ W,
    const float* __restrict__ bias, float* __restrict__ C,
    const float* __restrict__ aw_src, const float* __restrict__ aw_dst,
    float* __restrict__ o_asrc, float* __restrict__ o_adst,
    int nrows, int do_relu)
{
    constexpr int NT = NC / 16;
    constexpr int KS = K / 32;
    __shared__ __align__(32) _Float16 Bt[NC * K];       // W transposed: [n][k]
    __shared__ __align__(32) _Float16 As[8 * 16 * K];   // per-wave 16 rows

    const int tid = threadIdx.x, lane = tid & 31, wave = tid >> 5;
    const int rowbase = blockIdx.x * 128 + wave * 16;

    for (int i = tid; i < NC * K; i += 256) {
        int n = i / K, k = i - n * K;
        Bt[i] = (_Float16)W[k * NC + n];
    }
    for (int i = lane; i < 16 * K; i += 32) {
        int m = i / K, k = i - m * K;
        int r = rowbase + m;
        float v = (r < nrows) ? A[(size_t)r * K + k] : 0.0f;
        As[wave * 16 * K + i] = (_Float16)v;
    }
    __syncthreads();

    v8f acc[NT];
#pragma unroll
    for (int nt = 0; nt < NT; ++nt)
#pragma unroll
        for (int j = 0; j < 8; ++j) acc[nt][j] = 0.0f;

    const int m15 = lane & 15;
    const int ao = (lane & 16) ? 16 : 0;    // A half-offset per lane group
    const int bo = (lane & 16) ? 16 : 0;    // B half-offset per lane group

#pragma unroll
    for (int ks = 0; ks < KS; ++ks) {
        v16h af = make_afrag(&As[wave * 16 * K + m15 * K + ks * 32 + ao]);
#pragma unroll
        for (int nt = 0; nt < NT; ++nt) {
            int n = nt * 16 + m15;
            v16h bf = *(const v16h*)&Bt[n * K + ks * 32 + bo];
            acc[nt] = __builtin_amdgcn_wmma_f32_16x16x32_f16(
                false, af, false, bf, (short)0, acc[nt], false, false);
        }
    }

    const int group = lane >> 4, ncol = lane & 15;
#pragma unroll
    for (int nt = 0; nt < NT; ++nt) {
        int n = nt * 16 + ncol;
        float bv = bias ? bias[n] : 0.0f;
#pragma unroll
        for (int r = 0; r < 8; ++r) {
            int row = rowbase + r + 8 * group;
            if (row < nrows) {
                float v = acc[nt][r] + bv;
                if (do_relu) v = fmaxf(v, 0.0f);
                C[(size_t)row * NC + n] = v;
            }
        }
    }

    if (ALPHA) {
        float ps[8], pd[8];
#pragma unroll
        for (int r = 0; r < 8; ++r) { ps[r] = 0.f; pd[r] = 0.f; }
#pragma unroll
        for (int nt = 0; nt < NT; ++nt) {
            float was = aw_src[nt * 16 + ncol];
            float wad = aw_dst[nt * 16 + ncol];
#pragma unroll
            for (int r = 0; r < 8; ++r) { ps[r] += acc[nt][r] * was; pd[r] += acc[nt][r] * wad; }
        }
#pragma unroll
        for (int r = 0; r < 8; ++r) {
#pragma unroll
            for (int m = 1; m < 16; m <<= 1) {   // reduce within 16-lane half-wave
                ps[r] += __shfl_xor(ps[r], m, 32);
                pd[r] += __shfl_xor(pd[r], m, 32);
            }
        }
        if (ncol == 0) {
#pragma unroll
            for (int r = 0; r < 8; ++r) {
                int row = rowbase + r + 8 * group;
                if (row < nrows) { o_asrc[row] = ps[r]; o_adst[row] = pd[r]; }
            }
        }
    }
}

// ---------------------------------------------------------------------------
// Per-layer init: amax = enc(-inf), den = 0, h_out = 0
// ---------------------------------------------------------------------------
__global__ void gat_init(unsigned* __restrict__ amax, float* __restrict__ den,
                         float* __restrict__ hout, int n)
{
    int i = blockIdx.x * blockDim.x + threadIdx.x;
    if (i < n) { amax[i] = 0x007FFFFFu; den[i] = 0.0f; }  // enc(-inf)
    if (i < n * 64) hout[i] = 0.0f;
}

// ---------------------------------------------------------------------------
// Edge pass 1: segment-max of leaky-relu logits (order-insensitive via enc)
// ---------------------------------------------------------------------------
__global__ void edge_max(const int* __restrict__ src, const int* __restrict__ dst,
                         const float* __restrict__ asrc, const float* __restrict__ adst,
                         unsigned* __restrict__ amax, int E, int E2)
{
    int e = blockIdx.x * blockDim.x + threadIdx.x;
    if (e >= E2) return;
    int s, d;
    if (e < E) { s = src[e]; d = dst[e]; } else { s = d = e - E; }
    float lg = lrelu02(asrc[s] + adst[d]);
    atomicMax(&amax[d], encf(lg));
}

// ---------------------------------------------------------------------------
// Edge pass 2: ex = exp(logit - max), den[d] += ex
// ---------------------------------------------------------------------------
__global__ void edge_exp(const int* __restrict__ src, const int* __restrict__ dst,
                         const float* __restrict__ asrc, const float* __restrict__ adst,
                         const unsigned* __restrict__ amax, float* __restrict__ den,
                         float* __restrict__ exbuf, int E, int E2)
{
    int e = blockIdx.x * blockDim.x + threadIdx.x;
    if (e >= E2) return;
    int s, d;
    if (e < E) { s = src[e]; d = dst[e]; } else { s = d = e - E; }
    float lg = lrelu02(asrc[s] + adst[d]);
    float ex = expf(lg - decf(amax[d]));
    exbuf[e] = ex;
    atomAddF(&den[d], ex);
}

// ---------------------------------------------------------------------------
// Edge pass 3: out[d] += hW[s] * coef  (16 threads/edge, 4 channels each)
// ---------------------------------------------------------------------------
__global__ void edge_aggr(const int* __restrict__ src, const int* __restrict__ dst,
                          const float* __restrict__ hW, const float* __restrict__ exbuf,
                          const float* __restrict__ den, float* __restrict__ out,
                          int E, int E2)
{
    int t = blockIdx.x * blockDim.x + threadIdx.x;
    int e = t >> 4;
    if (e >= E2) return;
    int c0 = (t & 15) * 4;
    int s, d;
    if (e < E) { s = src[e]; d = dst[e]; } else { s = d = e - E; }
    float coef = exbuf[e] / (den[d] + 1e-16f);
    const float4 v = *(const float4*)(hW + (size_t)s * 64 + c0);
    float* op = out + (size_t)d * 64 + c0;
    atomAddF(op + 0, v.x * coef);
    atomAddF(op + 1, v.y * coef);
    atomAddF(op + 2, v.z * coef);
    atomAddF(op + 3, v.w * coef);
}

// ---------------------------------------------------------------------------
// h += bias (per 64-wide row), optional relu (in place)
// ---------------------------------------------------------------------------
__global__ void bias_relu(float* __restrict__ h, const float* __restrict__ b,
                          int total, int do_relu)
{
    int i = blockIdx.x * blockDim.x + threadIdx.x;
    if (i >= total) return;
    float v = h[i] + b[i & 63];
    if (do_relu) v = fmaxf(v, 0.0f);
    h[i] = v;
}

// ---------------------------------------------------------------------------
// Final edge MLP: t = [h3[src]|h3[dst]] (128), out = relu(t@Wm1+bm1)@Wm2+bm2
// WMMA GEMM (M=16 edges/wave, N=32, K=128) with fused 2nd-layer reduction.
// ---------------------------------------------------------------------------
__global__ __launch_bounds__(256) void mlp_edge(
    const float* __restrict__ h3, const int* __restrict__ src, const int* __restrict__ dst,
    const float* __restrict__ Wm1, const float* __restrict__ bm1,
    const float* __restrict__ Wm2, const float* __restrict__ bm2,
    float* __restrict__ out, int nedges)
{
    __shared__ __align__(32) _Float16 Bt[32 * 128];     // Wm1 transposed [n][k]
    __shared__ __align__(32) _Float16 As[8 * 16 * 128]; // per-wave 16 edge rows

    const int tid = threadIdx.x, lane = tid & 31, wave = tid >> 5;
    const int ebase = blockIdx.x * 128 + wave * 16;

    for (int i = tid; i < 32 * 128; i += 256) {
        int n = i >> 7, k = i & 127;
        Bt[i] = (_Float16)Wm1[k * 32 + n];
    }
    for (int i = lane; i < 16 * 128; i += 32) {
        int el = i >> 7, c = i & 127;
        int e = ebase + el;
        float v = 0.0f;
        if (e < nedges) {
            int node = (c < 64) ? src[e] : dst[e];
            v = h3[(size_t)node * 64 + (c & 63)];
        }
        As[wave * 16 * 128 + i] = (_Float16)v;
    }
    __syncthreads();

    v8f acc0, acc1;
#pragma unroll
    for (int j = 0; j < 8; ++j) { acc0[j] = 0.0f; acc1[j] = 0.0f; }

    const int m15 = lane & 15;
    const int ao = (lane & 16) ? 16 : 0;
    const int bo = (lane & 16) ? 16 : 0;

#pragma unroll
    for (int ks = 0; ks < 4; ++ks) {
        v16h af = make_afrag(&As[wave * 16 * 128 + m15 * 128 + ks * 32 + ao]);
        v16h b0 = *(const v16h*)&Bt[(m15)      * 128 + ks * 32 + bo];
        v16h b1 = *(const v16h*)&Bt[(16 + m15) * 128 + ks * 32 + bo];
        acc0 = __builtin_amdgcn_wmma_f32_16x16x32_f16(false, af, false, b0, (short)0, acc0, false, false);
        acc1 = __builtin_amdgcn_wmma_f32_16x16x32_f16(false, af, false, b1, (short)0, acc1, false, false);
    }

    const int group = lane >> 4, ncol = lane & 15;
    const float b1a = bm1[ncol], b1b = bm1[16 + ncol];
    const float w2a = Wm2[ncol], w2b = Wm2[16 + ncol];
    const float bo2 = bm2[0];

    float p[8];
#pragma unroll
    for (int r = 0; r < 8; ++r)
        p[r] = fmaxf(acc0[r] + b1a, 0.0f) * w2a + fmaxf(acc1[r] + b1b, 0.0f) * w2b;
#pragma unroll
    for (int r = 0; r < 8; ++r)
#pragma unroll
        for (int m = 1; m < 16; m <<= 1) p[r] += __shfl_xor(p[r], m, 32);

    if (ncol == 0) {
#pragma unroll
        for (int r = 0; r < 8; ++r) {
            int e = ebase + r + 8 * group;
            if (e < nedges) out[e] = p[r] + bo2;
        }
    }
}

// ---------------------------------------------------------------------------
extern "C" void kernel_launch(void* const* d_in, const int* in_sizes, int n_in,
                              void* d_out, int out_size, void* d_ws, size_t ws_size,
                              hipStream_t stream)
{
    const float* x   = (const float*)d_in[0];
    const int*   ei  = (const int*)  d_in[1];
    const float* W0  = (const float*)d_in[2];  const float* b0  = (const float*)d_in[3];
    const float* W1  = (const float*)d_in[4];  const float* as1 = (const float*)d_in[5];
    const float* ad1 = (const float*)d_in[6];  const float* b1  = (const float*)d_in[7];
    const float* W2  = (const float*)d_in[8];  const float* as2 = (const float*)d_in[9];
    const float* ad2 = (const float*)d_in[10]; const float* b2  = (const float*)d_in[11];
    const float* W3  = (const float*)d_in[12]; const float* as3 = (const float*)d_in[13];
    const float* ad3 = (const float*)d_in[14]; const float* b3  = (const float*)d_in[15];
    const float* Wm1 = (const float*)d_in[16]; const float* bm1 = (const float*)d_in[17];
    const float* Wm2 = (const float*)d_in[18]; const float* bm2 = (const float*)d_in[19];

    const int N  = in_sizes[0] / 128;
    const int E  = in_sizes[1] / 2;
    const int E2 = E + N;
    const int* src = ei;
    const int* dst = ei + E;

    float* ws = (float*)d_ws;
    float*    nodeA = ws;                                  // N*64
    float*    nodeB = nodeA + (size_t)N * 64;              // N*64 (hW)
    float*    nodeC = nodeB + (size_t)N * 64;              // N*64
    float*    asrc  = nodeC + (size_t)N * 64;              // N
    float*    adst  = asrc + N;                            // N
    unsigned* amax  = (unsigned*)(adst + N);               // N
    float*    den   = (float*)(amax + N);                  // N
    float*    exbuf = den + N;                             // E2

    const int TB = 256;
    const int gemm_blocks = (N + 127) / 128;
    const int nwide_blocks = (N * 64 + TB - 1) / TB;
    const int edge_blocks = (E2 + TB - 1) / TB;
    const int aggr_blocks = (E2 * 16 + TB - 1) / TB;

    // h0 = x @ W0 + b0                                  [N,128]x[128,32]
    gemm_tiled<128, 32, false><<<gemm_blocks, TB, 0, stream>>>(
        x, W0, b0, nodeA, nullptr, nullptr, nullptr, nullptr, N, 0);

    // ---- GAT layer 1: in=nodeA(K=32), hW=nodeB, out=nodeC ----
    gemm_tiled<32, 64, true><<<gemm_blocks, TB, 0, stream>>>(
        nodeA, W1, nullptr, nodeB, as1, ad1, asrc, adst, N, 0);
    gat_init<<<nwide_blocks, TB, 0, stream>>>(amax, den, nodeC, N);
    edge_max<<<edge_blocks, TB, 0, stream>>>(src, dst, asrc, adst, amax, E, E2);
    edge_exp<<<edge_blocks, TB, 0, stream>>>(src, dst, asrc, adst, amax, den, exbuf, E, E2);
    edge_aggr<<<aggr_blocks, TB, 0, stream>>>(src, dst, nodeB, exbuf, den, nodeC, E, E2);
    bias_relu<<<nwide_blocks, TB, 0, stream>>>(nodeC, b1, N * 64, 1);

    // ---- GAT layer 2: in=nodeC, hW=nodeB, out=nodeA ----
    gemm_tiled<64, 64, true><<<gemm_blocks, TB, 0, stream>>>(
        nodeC, W2, nullptr, nodeB, as2, ad2, asrc, adst, N, 0);
    gat_init<<<nwide_blocks, TB, 0, stream>>>(amax, den, nodeA, N);
    edge_max<<<edge_blocks, TB, 0, stream>>>(src, dst, asrc, adst, amax, E, E2);
    edge_exp<<<edge_blocks, TB, 0, stream>>>(src, dst, asrc, adst, amax, den, exbuf, E, E2);
    edge_aggr<<<aggr_blocks, TB, 0, stream>>>(src, dst, nodeB, exbuf, den, nodeA, E, E2);
    bias_relu<<<nwide_blocks, TB, 0, stream>>>(nodeA, b2, N * 64, 1);

    // ---- GAT layer 3 (no relu): in=nodeA, hW=nodeB, out=nodeC ----
    gemm_tiled<64, 64, true><<<gemm_blocks, TB, 0, stream>>>(
        nodeA, W3, nullptr, nodeB, as3, ad3, asrc, adst, N, 0);
    gat_init<<<nwide_blocks, TB, 0, stream>>>(amax, den, nodeC, N);
    edge_max<<<edge_blocks, TB, 0, stream>>>(src, dst, asrc, adst, amax, E, E2);
    edge_exp<<<edge_blocks, TB, 0, stream>>>(src, dst, asrc, adst, amax, den, exbuf, E, E2);
    edge_aggr<<<aggr_blocks, TB, 0, stream>>>(src, dst, nodeB, exbuf, den, nodeC, E, E2);
    bias_relu<<<nwide_blocks, TB, 0, stream>>>(nodeC, b3, N * 64, 0);

    // ---- final per-edge MLP (original E edges only) ----
    mlp_edge<<<(E + 127) / 128, TB, 0, stream>>>(
        nodeC, src, dst, Wm1, bm1, Wm2, bm2, (float*)d_out, E);
}
